// BinaryLinear_75024488726578
// MI455X (gfx1250) — matrix operand order repaired and checked by reading, not compiled, not run on hardware
//
#include <hip/hip_runtime.h>

// ---------------------------------------------------------------------------
// BinaryLinear: y = x @ sign(W)^T + b      (M=8192, N=4096, K=4096)
//
// Two-pass plan (MI455X reasoning):
//  - Pass 1: x -> bf16, sign(W) -> bf16 (exact in bf16) into d_ws (96 MB).
//    Shrinks GEMM working set from 192 MB (== L2) to 96 MB (L2-resident),
//    and removes all conversion VALU from the GEMM inner loop.
//  - Pass 2: bf16 WMMA GEMM, 128x128 block tile, KT=64, double-buffered LDS
//    staged with GLOBAL_LOAD_ASYNC_TO_LDS_B128 (ASYNCcnt) — builtin exists on
//    this toolchain; params are (AS1 i32x4*, AS3 i32x4*, imm, imm).
//  - If ws_size < 96 MB: single-pass fused kernel (convert while staging).
// ---------------------------------------------------------------------------

typedef __attribute__((ext_vector_type(16))) __bf16 v16bf;
typedef __attribute__((ext_vector_type(8)))  __bf16 v8bf;
typedef __attribute__((ext_vector_type(4)))  __bf16 v4bf;
typedef __attribute__((ext_vector_type(8)))  float  v8f;
typedef __attribute__((ext_vector_type(4)))  int    i32x4;

#define M_DIM 8192
#define N_DIM 4096
#define K_DIM 4096
#define MT 128
#define NT 128
#define KT2 64          // k-chunk for the bf16 GEMM (2 WMMA k-steps)
#define LDA 72          // LDS row stride in bf16 elems (64 + 8 pad = 144B)

__device__ __forceinline__ __bf16 f2bf(float f) {
  unsigned u = __builtin_bit_cast(unsigned, f);
  u += 0x7FFFu + ((u >> 16) & 1u);          // RNE
  unsigned short h = (unsigned short)(u >> 16);
  return __builtin_bit_cast(__bf16, h);
}

__device__ __forceinline__ __bf16 signbf(float f) {
  unsigned short h = (f > 0.f) ? (unsigned short)0x3F80u
                               : ((f < 0.f) ? (unsigned short)0xBF80u
                                            : (unsigned short)0x0000u);
  return __builtin_bit_cast(__bf16, h);
}

// ---------------------------------------------------------------------------
// Async copy plumbing (guarded: falls back to sync load + ds_store)
// ---------------------------------------------------------------------------
#if __has_builtin(__builtin_amdgcn_global_load_async_to_lds_b128)
#define HAVE_ASYNC_LDS 1
__device__ __forceinline__ void cp16_async(void* g, void* l) {
  __builtin_amdgcn_global_load_async_to_lds_b128(
      (__attribute__((address_space(1))) i32x4*)g,
      (__attribute__((address_space(3))) i32x4*)l, 0, 0);
}
#else
#define HAVE_ASYNC_LDS 0
__device__ __forceinline__ void cp16_async(void* g, void* l) {
  *(uint4*)l = *(const uint4*)g;            // global_load_b128 + ds_store_b128
}
#endif

template <int N>
__device__ __forceinline__ void wait_async() {
#if HAVE_ASYNC_LDS
#if __has_builtin(__builtin_amdgcn_s_wait_asynccnt)
  __builtin_amdgcn_s_wait_asynccnt((short)N);
#else
  asm volatile("s_wait_asynccnt %0" ::"i"(N) : "memory");
#endif
#endif
}

// ---------------------------------------------------------------------------
// Pass 1: conversion kernels
// ---------------------------------------------------------------------------
__global__ __launch_bounds__(256)
void convert_x_kernel(const float* __restrict__ x, __bf16* __restrict__ xb) {
  const size_t i = (size_t)blockIdx.x * 256 + threadIdx.x;   // float4 index
  float4 v = ((const float4*)x)[i];
  v4bf o;
  o[0] = f2bf(v.x); o[1] = f2bf(v.y); o[2] = f2bf(v.z); o[3] = f2bf(v.w);
  ((v4bf*)xb)[i] = o;
}

__global__ __launch_bounds__(256)
void convert_w_kernel(const float* __restrict__ w, __bf16* __restrict__ wb) {
  const size_t i = (size_t)blockIdx.x * 256 + threadIdx.x;
  float4 v = ((const float4*)w)[i];
  v4bf o;
  o[0] = signbf(v.x); o[1] = signbf(v.y); o[2] = signbf(v.z); o[3] = signbf(v.w);
  ((v4bf*)wb)[i] = o;
}

// ---------------------------------------------------------------------------
// Pass 2: bf16 WMMA GEMM, double-buffered async staging
// ---------------------------------------------------------------------------
__global__ __launch_bounds__(256)
void gemm_bf16_kernel(const __bf16* __restrict__ xb,
                      const __bf16* __restrict__ wb,
                      const float* __restrict__ bias,
                      float* __restrict__ out) {
  __shared__ alignas(16) __bf16 Abuf[2][MT][LDA];
  __shared__ alignas(16) __bf16 Bbuf[2][NT][LDA];

  const int tid   = threadIdx.x;
  const int lane  = tid & 31;
  const int wv    = tid >> 5;
  const int waveM = wv & 1;                 // 2 waves x 64 rows
  const int waveN = wv >> 1;                // 4 waves x 32 cols
  const int l16   = lane & 15;
  const int lHigh = lane >> 4;

  const int nBlock = blockIdx.x * NT;
  const int mBlock = blockIdx.y * MT;

  v8f acc[4][2];
  const v8f vzero = {};
#pragma unroll
  for (int fm = 0; fm < 4; ++fm)
#pragma unroll
    for (int fn = 0; fn < 2; ++fn) acc[fm][fn] = vzero;

  // Staging: 128 rows x 64 bf16 = 8 chunks of 16B per row; 1024 chunks/tile;
  // 4 A-chunks + 4 B-chunks per thread => 8 async ops per wave per stage.
  const int r0 = tid >> 3;                  // chunk row for i=0 (0..31)
  const int c0 = (tid & 7) << 3;            // bf16 col offset (0,8,...,56)

#define STAGE(buf, k0)                                                        \
  _Pragma("unroll")                                                           \
  for (int i = 0; i < 4; ++i) {                                               \
    const int r = r0 + i * 32;                                                \
    cp16_async((void*)&xb[(size_t)(mBlock + r) * K_DIM + (k0) + c0],          \
               (void*)&Abuf[buf][r][c0]);                                     \
    cp16_async((void*)&wb[(size_t)(nBlock + r) * K_DIM + (k0) + c0],          \
               (void*)&Bbuf[buf][r][c0]);                                     \
  }

  STAGE(0, 0)

  const int NKT = K_DIM / KT2;              // 64 iterations
  for (int kt = 0; kt < NKT; ++kt) {
    const int cur = kt & 1;

    if (kt + 1 < NKT) {
      STAGE(cur ^ 1, (kt + 1) * KT2)
      wait_async<8>();                      // oldest 8 (== cur) complete
    } else {
      wait_async<0>();
    }
    __syncthreads();                        // cur tile visible to all waves

#pragma unroll
    for (int s = 0; s < 2; ++s) {           // two k=32 WMMA steps
      const int kc = s * 32;

      v16bf afrag[4];
#pragma unroll
      for (int fm = 0; fm < 4; ++fm) {
        const int m = waveM * 64 + fm * 16 + l16;
        const v8bf lo = *(const v8bf*)&Abuf[cur][m][kc + lHigh * 8];
        const v8bf hi = *(const v8bf*)&Abuf[cur][m][kc + lHigh * 8 + 16];
        v16bf a;
#pragma unroll
        for (int e = 0; e < 8; ++e) { a[e] = lo[e]; a[e + 8] = hi[e]; }
        afrag[fm] = a;
      }

      v16bf bfrag[2];
#pragma unroll
      for (int fn = 0; fn < 2; ++fn) {
        const int n = waveN * 32 + fn * 16 + l16;
        const v8bf lo = *(const v8bf*)&Bbuf[cur][n][kc + lHigh * 8];
        const v8bf hi = *(const v8bf*)&Bbuf[cur][n][kc + lHigh * 8 + 16];
        v16bf b;
#pragma unroll
        for (int e = 0; e < 8; ++e) { b[e] = lo[e]; b[e + 8] = hi[e]; }
        bfrag[fn] = b;
      }

#pragma unroll
      for (int fm = 0; fm < 4; ++fm)
#pragma unroll
        for (int fn = 0; fn < 2; ++fn)
          acc[fm][fn] = __builtin_amdgcn_wmma_f32_16x16x32_bf16(
              false, afrag[fm], false, bfrag[fn],
              (short)0, acc[fm][fn], false, false);
    }
    __syncthreads();                        // all reads of cur done before
  }                                         // it is overwritten next iter

#pragma unroll
  for (int fn = 0; fn < 2; ++fn) {
    const int n = nBlock + waveN * 32 + fn * 16 + l16;
    const float bv = bias[n];
#pragma unroll
    for (int fm = 0; fm < 4; ++fm) {
      const int mbase = mBlock + waveM * 64 + fm * 16 + lHigh * 8;
#pragma unroll
      for (int e = 0; e < 8; ++e)
        out[(size_t)(mbase + e) * N_DIM + n] = acc[fm][fn][e] + bv;
    }
  }
#undef STAGE
}

// ---------------------------------------------------------------------------
// Fallback: single-pass fused kernel (convert while staging), KT=32
// ---------------------------------------------------------------------------
#define KT1 32
#define LDSS 40

__global__ __launch_bounds__(256)
void binlin_fused_kernel(const float* __restrict__ x,
                         const float* __restrict__ w,
                         const float* __restrict__ bias,
                         float* __restrict__ out) {
  __shared__ alignas(16) __bf16 As[MT][LDSS];
  __shared__ alignas(16) __bf16 Bs[NT][LDSS];

  const int tid   = threadIdx.x;
  const int lane  = tid & 31;
  const int wv    = tid >> 5;
  const int waveM = wv & 1;
  const int waveN = wv >> 1;
  const int l16   = lane & 15;
  const int lHigh = lane >> 4;

  const int nBlock = blockIdx.x * NT;
  const int mBlock = blockIdx.y * MT;

  v8f acc[4][2];
  const v8f vzero = {};
#pragma unroll
  for (int fm = 0; fm < 4; ++fm)
#pragma unroll
    for (int fn = 0; fn < 2; ++fn) acc[fm][fn] = vzero;

  const float4* xv  = (const float4*)x;
  const float4* wv4 = (const float4*)w;

  for (int kt = 0; kt < K_DIM / KT1; ++kt) {
    const int k0 = kt * KT1;
    __syncthreads();
#pragma unroll
    for (int i = 0; i < 4; ++i) {
      const int idx = tid + i * 256;
      const int r   = idx >> 3;
      const int c4  = (idx & 7) << 2;
      float4 va = xv[(((size_t)(mBlock + r)) * K_DIM + (size_t)(k0 + c4)) >> 2];
      As[r][c4 + 0] = f2bf(va.x); As[r][c4 + 1] = f2bf(va.y);
      As[r][c4 + 2] = f2bf(va.z); As[r][c4 + 3] = f2bf(va.w);
      float4 vb = wv4[(((size_t)(nBlock + r)) * K_DIM + (size_t)(k0 + c4)) >> 2];
      Bs[r][c4 + 0] = signbf(vb.x); Bs[r][c4 + 1] = signbf(vb.y);
      Bs[r][c4 + 2] = signbf(vb.z); Bs[r][c4 + 3] = signbf(vb.w);
    }
    __syncthreads();

    v16bf afrag[4];
#pragma unroll
    for (int fm = 0; fm < 4; ++fm) {
      const int m = waveM * 64 + fm * 16 + l16;
      const v8bf lo = *(const v8bf*)&As[m][lHigh * 8];
      const v8bf hi = *(const v8bf*)&As[m][lHigh * 8 + 16];
      v16bf a;
#pragma unroll
      for (int e = 0; e < 8; ++e) { a[e] = lo[e]; a[e + 8] = hi[e]; }
      afrag[fm] = a;
    }
    v16bf bfrag[2];
#pragma unroll
    for (int fn = 0; fn < 2; ++fn) {
      const int n = waveN * 32 + fn * 16 + l16;
      const v8bf lo = *(const v8bf*)&Bs[n][lHigh * 8];
      const v8bf hi = *(const v8bf*)&Bs[n][lHigh * 8 + 16];
      v16bf b;
#pragma unroll
      for (int e = 0; e < 8; ++e) { b[e] = lo[e]; b[e + 8] = hi[e]; }
      bfrag[fn] = b;
    }
#pragma unroll
    for (int fm = 0; fm < 4; ++fm)
#pragma unroll
      for (int fn = 0; fn < 2; ++fn)
        acc[fm][fn] = __builtin_amdgcn_wmma_f32_16x16x32_bf16(
            false, afrag[fm], false, bfrag[fn],
            (short)0, acc[fm][fn], false, false);
  }

#pragma unroll
  for (int fn = 0; fn < 2; ++fn) {
    const int n = nBlock + waveN * 32 + fn * 16 + l16;
    const float bv = bias[n];
#pragma unroll
    for (int fm = 0; fm < 4; ++fm) {
      const int mbase = mBlock + waveM * 64 + fm * 16 + lHigh * 8;
#pragma unroll
      for (int e = 0; e < 8; ++e)
        out[(size_t)(mbase + e) * N_DIM + n] = acc[fm][fn][e] + bv;
    }
  }
}

// ---------------------------------------------------------------------------
extern "C" void kernel_launch(void* const* d_in, const int* in_sizes, int n_in,
                              void* d_out, int out_size, void* d_ws, size_t ws_size,
                              hipStream_t stream) {
  (void)in_sizes; (void)n_in; (void)out_size;
  const float* x    = (const float*)d_in[0];   // [8,1024,4096]
  const float* w    = (const float*)d_in[1];   // [4096,4096]
  const float* bias = (const float*)d_in[2];   // [4096]
  float* out        = (float*)d_out;

  const size_t xElems = (size_t)M_DIM * K_DIM;           // 33,554,432
  const size_t wElems = (size_t)N_DIM * K_DIM;           // 16,777,216
  const size_t need   = (xElems + wElems) * sizeof(__bf16);  // 96 MB

  dim3 grid(N_DIM / NT, M_DIM / MT);                     // 32 x 64
  dim3 block(256);

  if (ws_size >= need) {
    __bf16* xb = (__bf16*)d_ws;
    __bf16* wb = xb + xElems;
    convert_x_kernel<<<dim3((unsigned)(xElems / 4 / 256)), block, 0, stream>>>(x, xb);
    convert_w_kernel<<<dim3((unsigned)(wElems / 4 / 256)), block, 0, stream>>>(w, wb);
    gemm_bf16_kernel<<<grid, block, 0, stream>>>(xb, wb, bias, out);
  } else {
    binlin_fused_kernel<<<grid, block, 0, stream>>>(x, w, bias, out);
  }
}
